// LEFTNetZ_60541859004790
// MI455X (gfx1250) — matrix operand order repaired and checked by reading, not compiled, not run on hardware
//
#include <hip/hip_runtime.h>
#include <math.h>

#define HID   128
#define NRBF  96
#define NL    4
#define NATOM 95
#define NNODE 10000
#define NEDGE 160000
#define NMOL  64
#define CUTOFF 6.0f
#define EPSF  1e-10f
#define ISQ2  0.70710678118654752f
#define ISQ3  0.57735026918962576f
#define ISQH  0.08838834764831845f   /* 1/sqrt(128) */
#define PI_F  3.14159265358979323f

typedef __attribute__((ext_vector_type(16))) __bf16 v16bf;
typedef __attribute__((ext_vector_type(8)))  float  v8f;

__device__ __forceinline__ float silu_f(float x) {
  // x * sigmoid(x); v_rcp_f32 instead of the IEEE divide expansion
  return x * __builtin_amdgcn_rcpf(1.f + __expf(-x));
}

// ---------------------------------------------------------------------------
// Generic GEMM: C[M,N] = act(A[M,K] @ W[N,K]^T + bias) * rowScale
// fp32 in memory, bf16 in-register (hardware cvt), v_wmma_f32_16x16x32_bf16.
// M%16==0, N%64==0, K%32==0, all row strides multiples of 4 floats (16B) --
// true for every matmul in this model, so all loads are b128.
// Each wave computes a 16x64 strip: one A fragment per k-step feeds 4 WMMAs
// (4x less A traffic; B = weights are L2-resident). 4 waves per block tile M.
// Fragment layouts per CDNA5 ISA 7.12.2:
//   A 16x32 bf16: lane m=l&15; group g=l>>4; halves[0..7]->K=k0+8g+i,
//                 halves[8..15]->K=k0+16+8g+i
//   B 32x16 bf16: lane n=l&15; same K mapping as A  (row n of W^T)
//   C/D: lane n=l&15; vgpr r -> M = m0 + r + 8g
// ---------------------------------------------------------------------------
#define NT 4   // n-tiles per wave

__device__ __forceinline__ v16bf cvt16(float4 f0, float4 f1, float4 f2, float4 f3) {
  v16bf v;
  v[0]  = (__bf16)f0.x; v[1]  = (__bf16)f0.y; v[2]  = (__bf16)f0.z; v[3]  = (__bf16)f0.w;
  v[4]  = (__bf16)f1.x; v[5]  = (__bf16)f1.y; v[6]  = (__bf16)f1.z; v[7]  = (__bf16)f1.w;
  v[8]  = (__bf16)f2.x; v[9]  = (__bf16)f2.y; v[10] = (__bf16)f2.z; v[11] = (__bf16)f2.w;
  v[12] = (__bf16)f3.x; v[13] = (__bf16)f3.y; v[14] = (__bf16)f3.z; v[15] = (__bf16)f3.w;
  return v;
}

template<int ACT, bool BIASF, bool RSF>
__global__ void gemm_wmma_k(const float* __restrict__ A, long lda,
                            const float* __restrict__ W,
                            const float* __restrict__ bias,
                            const float* __restrict__ rowScale,
                            float* __restrict__ C, long ldc,
                            int M, int N, int K)
{
  const int wave  = threadIdx.x >> 5;
  const int lane  = threadIdx.x & 31;
  const int m0    = (blockIdx.x * 4 + wave) * 16;
  if (m0 >= M) return;                          // wave-uniform: EXEC stays full
  const int n0    = blockIdx.y * (16 * NT);
  const int l15   = lane & 15;
  const int grp   = lane >> 4;

  const float* arow = A + (long)(m0 + l15) * lda;
  const float* wbase = W + (long)(n0 + l15) * K + grp * 8;

  v8f acc[NT];
#pragma unroll
  for (int t = 0; t < NT; ++t) acc[t] = (v8f){};

  for (int k0 = 0; k0 < K; k0 += 32) {
    const float* ap = arow + k0 + grp * 8;
    float4 a0 = *(const float4*)(ap);
    float4 a1 = *(const float4*)(ap + 4);
    float4 a2 = *(const float4*)(ap + 16);
    float4 a3 = *(const float4*)(ap + 20);
    v16bf av = cvt16(a0, a1, a2, a3);
#pragma unroll
    for (int t = 0; t < NT; ++t) {
      const float* bp = wbase + (long)(t * 16) * K + k0;
      float4 b0 = *(const float4*)(bp);
      float4 b1 = *(const float4*)(bp + 4);
      float4 b2 = *(const float4*)(bp + 16);
      float4 b3 = *(const float4*)(bp + 20);
      v16bf bv = cvt16(b0, b1, b2, b3);
      acc[t] = __builtin_amdgcn_wmma_f32_16x16x32_bf16(
          false, av, false, bv, (short)0, acc[t], false, false);
    }
  }

  float rs[8];
  if (RSF) {
#pragma unroll
    for (int r = 0; r < 8; ++r) rs[r] = rowScale[m0 + r + grp * 8];
  }
#pragma unroll
  for (int t = 0; t < NT; ++t) {
    const int col = n0 + t * 16 + l15;
    const float bb = BIASF ? bias[col] : 0.f;
#pragma unroll
    for (int r = 0; r < 8; ++r) {
      int mm = m0 + r + grp * 8;
      float v = acc[t][r] + bb;
      if (ACT == 1) v = silu_f(v);
      if (RSF) v *= rs[r];
      C[(long)mm * ldc + col] = v;
    }
  }
}

static inline void gemm(hipStream_t st, const float* A, long lda, const float* W,
                        const float* bias, const float* rs, float* C, long ldc,
                        int M, int N, int K, int act)
{
  dim3 g((M / 16 + 3) / 4, N / (16 * NT)), b(128);
  if (act) {
    gemm_wmma_k<1, true, false><<<g, b, 0, st>>>(A, lda, W, bias, rs, C, ldc, M, N, K);
  } else if (rs) {
    gemm_wmma_k<0, true, true><<<g, b, 0, st>>>(A, lda, W, bias, rs, C, ldc, M, N, K);
  } else if (bias) {
    gemm_wmma_k<0, true, false><<<g, b, 0, st>>>(A, lda, W, bias, rs, C, ldc, M, N, K);
  } else {
    gemm_wmma_k<0, false, false><<<g, b, 0, st>>>(A, lda, W, bias, rs, C, ldc, M, N, K);
  }
}

// ---------------------------------------------------------------------------
// LayerNorm over HID=128 (optional affine, optional trailing silu).
// One wave32 per row, 4 elements per lane, shuffle-xor reduction.
// ---------------------------------------------------------------------------
template<int ACT>
__global__ void layernorm_k(const float* __restrict__ x,
                            const float* __restrict__ g,
                            const float* __restrict__ b,
                            float* __restrict__ y, int n)
{
  int wave = threadIdx.x >> 5, lane = threadIdx.x & 31;
  int row = blockIdx.x * 4 + wave;
  if (row >= n) return;
  const float* xr = x + (long)row * HID;
  float v[4], s = 0.f, ss = 0.f;
#pragma unroll
  for (int i = 0; i < 4; ++i) { v[i] = xr[lane + 32 * i]; s += v[i]; ss += v[i] * v[i]; }
#pragma unroll
  for (int m = 16; m >= 1; m >>= 1) { s += __shfl_xor(s, m, 32); ss += __shfl_xor(ss, m, 32); }
  float mu  = s * (1.f / HID);
  float var = ss * (1.f / HID) - mu * mu;
  float inv = rsqrtf(var + 1e-5f);
  float* yr = y + (long)row * HID;
#pragma unroll
  for (int i = 0; i < 4; ++i) {
    int c = lane + 32 * i;
    float o = (v[i] - mu) * inv;
    if (g) o = o * g[c] + b[c];
    if (ACT == 1) o = silu_f(o);
    yr[c] = o;
  }
}

// ---------------------------------------------------------------------------
// Edge geometry + RBF embedding (written into edge_weight[:,384:480]).
// ---------------------------------------------------------------------------
__global__ void geom_k(const float* __restrict__ pos, const int* __restrict__ ei,
                       float* __restrict__ rbounds, float* __restrict__ ediff,
                       float* __restrict__ eframe, float* __restrict__ ew)
{
  int e = blockIdx.x * blockDim.x + threadIdx.x;
  if (e >= NEDGE) return;
  int jn = ei[e], in_ = ei[NEDGE + e];
  float pjx = pos[jn * 3 + 0], pjy = pos[jn * 3 + 1], pjz = pos[jn * 3 + 2];
  float pix = pos[in_ * 3 + 0], piy = pos[in_ * 3 + 1], piz = pos[in_ * 3 + 2];
  float vx = pjx - pix, vy = pjy - piy, vz = pjz - piz;
  float dist = sqrtf(vx * vx + vy * vy + vz * vz);
  float idd = 1.f / (dist + EPSF);
  float dx = vx * idd, dy = vy * idd, dz = vz * idd;
  // cross(pos[i], pos[j]), normalized
  float cx = piy * pjz - piz * pjy;
  float cy = piz * pjx - pix * pjz;
  float cz = pix * pjy - piy * pjx;
  float icn = 1.f / (sqrtf(cx * cx + cy * cy + cz * cz) + EPSF);
  cx *= icn; cy *= icn; cz *= icn;
  // vertical = cross(diff, cross)
  float ux = dy * cz - dz * cy;
  float uy = dz * cx - dx * cz;
  float uz = dx * cy - dy * cx;

  float rb = 0.5f * (cosf(dist * PI_F / CUTOFF) + 1.f);
  rbounds[e] = rb;
  float mask = (dist < CUTOFF) ? rb : 0.f;

  ediff[e * 3 + 0] = dx; ediff[e * 3 + 1] = dy; ediff[e * 3 + 2] = dz;
  float* fr = eframe + (long)e * 9;                 // [s][f]: f0=diff f1=cross f2=vert
  fr[0] = dx; fr[1] = cx; fr[2] = ux;
  fr[3] = dy; fr[4] = cy; fr[5] = uy;
  fr[6] = dz; fr[7] = cz; fr[8] = uz;

  const float start = expf(-CUTOFF);
  const float span  = 2.f / NRBF * (1.f - start);
  const float beta  = 1.f / (span * span);
  float ex = expf(-dist);
  float* re = ew + (long)e * 480 + 384;
#pragma unroll 8
  for (int r = 0; r < NRBF; ++r) {
    float mean = start + (1.f - start) * ((float)r / (NRBF - 1));
    float t = ex - mean;
    re[r] = mask * __expf(-beta * t * t);
  }
}

__global__ void embed_gather_k(const int* __restrict__ z, const float* __restrict__ W,
                               const float* __restrict__ b, float* __restrict__ out)
{
  long t = (long)blockIdx.x * blockDim.x + threadIdx.x;
  if (t >= (long)NNODE * HID) return;
  int n = (int)(t / HID), h = (int)(t % HID);
  out[t] = W[h * NATOM + z[n]] + b[h];
}

__global__ void zero_k(float* p, long n)
{ long t = (long)blockIdx.x * blockDim.x + threadIdx.x; if (t < n) p[t] = 0.f; }

__global__ void copy_k(float* __restrict__ d, const float* __restrict__ s, long n)
{ long t = (long)blockIdx.x * blockDim.x + threadIdx.x; if (t < n) d[t] = s[t]; }

// s += seg_i( radial_hidden * z_emb[j] )  (radial_hidden lives in ew[:,256:384])
__global__ void sinit_k(const int* __restrict__ ei, const float* __restrict__ ew,
                        const float* __restrict__ zemb, float* __restrict__ s)
{
  int e = blockIdx.x, c = threadIdx.x;
  int jn = ei[e], in_ = ei[NEDGE + e];
  float rh = ew[(long)e * 480 + 256 + c];
  atomicAdd(&s[(long)in_ * HID + c], rh * zemb[(long)jn * HID + c]);
}

// NE1[i] += radial_hidden * edge_diff[:,s] * s1[j]
__global__ void ne1_k(const int* __restrict__ ei, const float* __restrict__ ew,
                      const float* __restrict__ s1, const float* __restrict__ ediff,
                      float* __restrict__ ne1)
{
  int e = blockIdx.x, c = threadIdx.x;
  int jn = ei[e], in_ = ei[NEDGE + e];
  float coef = ew[(long)e * 480 + 256 + c] * s1[(long)jn * HID + c];
#pragma unroll
  for (int sd = 0; sd < 3; ++sd)
    atomicAdd(&ne1[((long)in_ * 3 + sd) * HID + c], ediff[e * 3 + sd] * coef);
}

// scalarize(NE1[i]), scalarize(NE1[j]) -> ew[:,0:128], ew[:,128:256] (scaled by rbounds)
__global__ void scalarize_k(const int* __restrict__ ei, const float* __restrict__ ne1,
                            const float* __restrict__ eframe, const float* __restrict__ rbounds,
                            const float* __restrict__ l0W, const float* __restrict__ l0b,
                            const float* __restrict__ l2W, const float* __restrict__ l2b,
                            float* __restrict__ ew)
{
  __shared__ float w0[96], b0[32], w2[32], fr[9];
  int e = blockIdx.x, h = threadIdx.x;
  if (h < 96) w0[h] = l0W[h];
  else if (h < 128) b0[h - 96] = l0b[h - 96];
  if (h < 32) w2[h] = l2W[h];
  if (h < 9)  fr[h] = eframe[(long)e * 9 + h];
  __syncthreads();

  int jn = ei[e], in_ = ei[NEDGE + e];
  float rb = rbounds[e], b2 = l2b[0];
#pragma unroll
  for (int which = 0; which < 2; ++which) {
    int node = which ? jn : in_;                 // scalar3 uses NE1[i], scalar4 NE1[j]
    float n0 = ne1[((long)node * 3 + 0) * HID + h];
    float n1 = ne1[((long)node * 3 + 1) * HID + h];
    float n2 = ne1[((long)node * 3 + 2) * HID + h];
    float sc0 = n0 * fr[0] + n1 * fr[3] + n2 * fr[6];
    float sc1 = fabsf(n0 * fr[1] + n1 * fr[4] + n2 * fr[7]);
    float sc2 = n0 * fr[2] + n1 * fr[5] + n2 * fr[8];
    float y = b2 + sc0;
#pragma unroll 4
    for (int o = 0; o < 32; ++o) {
      float u = w0[o * 3] * sc0 + w0[o * 3 + 1] * sc1 + w0[o * 3 + 2] * sc2 + b0[o];
      y += w2[o] * silu_f(u);
    }
    ew[(long)e * 480 + which * HID + h] = y * rb;
  }
}

// message: m = xh[j] * (rbf_out * dir2_out); scatter to sacc / vacc
__global__ void message_k(const int* __restrict__ ei, const float* __restrict__ xh,
                          const float* __restrict__ d2o, const float* __restrict__ rbo,
                          const float* __restrict__ ediff, const float* __restrict__ vecb,
                          float* __restrict__ sacc, float* __restrict__ vacc)
{
  int e = blockIdx.x, c = threadIdx.x;
  int jn = ei[e], in_ = ei[NEDGE + e];
  long eb = (long)e * 384, jb = (long)jn * 384;
  float r0 = rbo[eb + c]       * d2o[eb + c];
  float r1 = rbo[eb + 128 + c] * d2o[eb + 128 + c];
  float r2 = rbo[eb + 256 + c] * d2o[eb + 256 + c];
  float x1 = xh[jb + c]       * r0;
  float h2 = xh[jb + 128 + c] * r1 * ISQ3;
  float h3 = xh[jb + 256 + c] * r2;
  atomicAdd(&sacc[(long)in_ * HID + c], x1);
#pragma unroll
  for (int sd = 0; sd < 3; ++sd) {
    float vm = (vecb[((long)jn * 3 + sd) * HID + c] * h2 + h3 * ediff[e * 3 + sd]) * ISQH;
    atomicAdd(&vacc[((long)in_ * 3 + sd) * HID + c], vm);
  }
}

__global__ void nup1_k(float* __restrict__ s, const float* __restrict__ sacc,
                       float* __restrict__ vecb, const float* __restrict__ vacc)
{
  int n = blockIdx.x, c = threadIdx.x;
  long b = (long)n * HID;
  s[b + c] = (s[b + c] + sacc[b + c]) * ISQ2;
#pragma unroll
  for (int sd = 0; sd < 3; ++sd) { long o = ((long)n * 3 + sd) * HID + c; vecb[o] += vacc[o]; }
}

__global__ void makecat_k(const float* __restrict__ s, const float* __restrict__ vp,
                          float* __restrict__ cat, float* __restrict__ vdot)
{
  int n = blockIdx.x, c = threadIdx.x;
  float sq = 1e-10f, dt = 0.f;
#pragma unroll
  for (int sd = 0; sd < 3; ++sd) {
    long o = ((long)n * 3 + sd) * 256;
    float a = vp[o + c], b = vp[o + 128 + c];
    sq += a * a; dt += a * b;
  }
  cat[(long)n * 256 + c]       = s[(long)n * HID + c];
  cat[(long)n * 256 + 128 + c] = sqrtf(sq);
  vdot[(long)n * HID + c]      = dt * ISQH;
}

__global__ void nup2_k(float* __restrict__ s, float* __restrict__ vecb,
                       const float* __restrict__ xv, const float* __restrict__ vp,
                       const float* __restrict__ vdot)
{
  int n = blockIdx.x, c = threadIdx.x;
  long xb = (long)n * 384;
  s[(long)n * HID + c] += (xv[xb + c] + xv[xb + 128 + c] + vdot[(long)n * HID + c]) * ISQ2;
  float a3 = xv[xb + 256 + c];
#pragma unroll
  for (int sd = 0; sd < 3; ++sd)
    vecb[((long)n * 3 + sd) * HID + c] += a3 * vp[((long)n * 3 + sd) * 256 + 128 + c];
}

__global__ void out_k(const float* __restrict__ s, const float* __restrict__ lw,
                      const float* __restrict__ lb, const int* __restrict__ batch,
                      float* __restrict__ outAcc, float* __restrict__ cnt)
{
  int n = blockIdx.x * blockDim.x + threadIdx.x;
  if (n >= NNODE) return;
  float d = lb[0];
  const float* sr = s + (long)n * HID;
#pragma unroll 8
  for (int h = 0; h < HID; ++h) d += sr[h] * lw[h];
  atomicAdd(&outAcc[batch[n]], d);
  atomicAdd(&cnt[batch[n]], 1.f);
}

__global__ void fin_k(const float* __restrict__ outAcc, const float* __restrict__ cnt,
                      float* __restrict__ out)
{
  int m = threadIdx.x;
  if (m < NMOL) out[m] = outAcc[m] / fmaxf(cnt[m], 1.f);
}

// ---------------------------------------------------------------------------
extern "C" void kernel_launch(void* const* d_in, const int* in_sizes, int n_in,
                              void* d_out, int out_size, void* d_ws, size_t ws_size,
                              hipStream_t stream)
{
  (void)in_sizes; (void)n_in; (void)out_size; (void)ws_size;
  const float* pos   = (const float*)d_in[0];
  const int*   z     = (const int*)d_in[1];
  const int*   ei    = (const int*)d_in[2];   // [2, NEDGE]: row0 = j, row1 = i
  const int*   batch = (const int*)d_in[3];

  // Param leaves in dict insertion order of make_params()
  int pi = 4;
  const float* radial0W = (const float*)d_in[pi++]; const float* radial0b = (const float*)d_in[pi++];
  const float* radial2W = (const float*)d_in[pi++]; const float* radial2b = (const float*)d_in[pi++];
  const float* nefcW    = (const float*)d_in[pi++]; const float* nefcb    = (const float*)d_in[pi++];
  const float* svW      = (const float*)d_in[pi++]; const float* svb      = (const float*)d_in[pi++];
  const float* l0W      = (const float*)d_in[pi++]; const float* l0b      = (const float*)d_in[pi++];
  const float* l2W      = (const float*)d_in[pi++]; const float* l2b      = (const float*)d_in[pi++];
  const float* lastW    = (const float*)d_in[pi++]; const float* lastb    = (const float*)d_in[pi++];
  const float *dir0W[NL], *dir0b[NL], *dir2W[NL], *dir2b[NL], *xp0W[NL], *xp0b[NL],
              *xp2W[NL], *xp2b[NL], *rbfW[NL], *rbfb[NL], *lng[NL], *lnb[NL],
              *vpW[NL], *xv0W[NL], *xv0b[NL], *xv2W[NL], *xv2b[NL];
  for (int l = 0; l < NL; ++l) {
    dir0W[l] = (const float*)d_in[pi++]; dir0b[l] = (const float*)d_in[pi++];
    dir2W[l] = (const float*)d_in[pi++]; dir2b[l] = (const float*)d_in[pi++];
    xp0W[l]  = (const float*)d_in[pi++]; xp0b[l]  = (const float*)d_in[pi++];
    xp2W[l]  = (const float*)d_in[pi++]; xp2b[l]  = (const float*)d_in[pi++];
    rbfW[l]  = (const float*)d_in[pi++]; rbfb[l]  = (const float*)d_in[pi++];
    lng[l]   = (const float*)d_in[pi++]; lnb[l]   = (const float*)d_in[pi++];
    vpW[l]   = (const float*)d_in[pi++];                                   // bias=False
    xv0W[l]  = (const float*)d_in[pi++]; xv0b[l]  = (const float*)d_in[pi++];
    xv2W[l]  = (const float*)d_in[pi++]; xv2b[l]  = (const float*)d_in[pi++];
  }

  // Workspace carving (256B aligned slices)
  char* base = (char*)d_ws; size_t off = 0;
  auto alloc = [&](long nf) -> float* {
    float* p = (float*)(base + off);
    off += ((size_t)nf * 4 + 255) & ~(size_t)255;
    return p;
  };
  float* ew      = alloc((long)NEDGE * 480);   // [sc3|sc4|rhid|remb]
  float* et1     = alloc((long)NEDGE * 384);
  float* et2     = alloc((long)NEDGE * 384);
  float* et3     = alloc((long)NEDGE * 384);
  float* rbounds = alloc(NEDGE);
  float* ediff   = alloc((long)NEDGE * 3);
  float* eframe  = alloc((long)NEDGE * 9);
  float* zemb    = alloc((long)NNODE * HID);
  float* sbuf    = alloc((long)NNODE * HID);
  float* tnA     = alloc((long)NNODE * HID);
  float* tnB     = alloc((long)NNODE * HID);
  float* xh      = alloc((long)NNODE * 384);
  float* s1      = alloc((long)NNODE * HID);
  float* ne1     = alloc((long)NNODE * 384);
  float* sacc    = alloc((long)NNODE * HID);
  float* vecb    = alloc((long)NNODE * 384);
  float* vacc    = alloc((long)NNODE * 384);
  float* vpb     = alloc((long)NNODE * 768);
  float* catb    = alloc((long)NNODE * 256);
  float* vdotb   = alloc((long)NNODE * HID);
  float* xvb     = alloc((long)NNODE * 384);
  float* outAcc  = alloc(NMOL);
  float* cntb    = alloc(NMOL);

  auto zero = [&](float* p, long n) {
    zero_k<<<(unsigned)((n + 255) / 256), 256, 0, stream>>>(p, n);
  };

  // 1) Edge geometry + radial embedding
  geom_k<<<(NEDGE + 255) / 256, 256, 0, stream>>>(pos, ei, rbounds, ediff, eframe, ew);

  // 2) radial MLP: silu(radial0 @ remb) then radial2, scaled by rbounds -> ew[:,256:384]
  gemm(stream, ew + 384, 480, radial0W, radial0b, nullptr, et1, 128, NEDGE, 128, NRBF, 1);
  gemm(stream, et1, 128, radial2W, radial2b, rbounds, ew + 256, 480, NEDGE, 128, 128, 0);

  // 3) node embedding: ln(ne_fc one-hot) -> zemb; s = zemb + seg(rhid * zemb[j], i)
  embed_gather_k<<<(unsigned)(((long)NNODE * HID + 255) / 256), 256, 0, stream>>>(z, nefcW, nefcb, tnA);
  layernorm_k<0><<<(NNODE + 3) / 4, 128, 0, stream>>>(tnA, nullptr, nullptr, zemb, NNODE);
  copy_k<<<(unsigned)(((long)NNODE * HID + 255) / 256), 256, 0, stream>>>(sbuf, zemb, (long)NNODE * HID);
  sinit_k<<<NEDGE, 128, 0, stream>>>(ei, ew, zemb, sbuf);

  // 4) s1 = silu(ln(sv_lin @ s)); NE1 accumulation
  gemm(stream, sbuf, 128, svW, svb, nullptr, tnB, 128, NNODE, 128, 128, 0);
  layernorm_k<1><<<(NNODE + 3) / 4, 128, 0, stream>>>(tnB, nullptr, nullptr, s1, NNODE);
  zero(ne1, (long)NNODE * 384);
  ne1_k<<<NEDGE, 128, 0, stream>>>(ei, ew, s1, ediff, ne1);

  // 5) scalarize -> ew[:,0:256]
  scalarize_k<<<NEDGE, 128, 0, stream>>>(ei, ne1, eframe, rbounds, l0W, l0b, l2W, l2b, ew);

  // 6) message-passing layers
  zero(vecb, (long)NNODE * 384);
  for (int l = 0; l < NL; ++l) {
    layernorm_k<0><<<(NNODE + 3) / 4, 128, 0, stream>>>(sbuf, lng[l], lnb[l], tnA, NNODE);
    gemm(stream, tnA, 128, xp0W[l], xp0b[l], nullptr, tnB, 128, NNODE, 128, 128, 1);
    gemm(stream, tnB, 128, xp2W[l], xp2b[l], nullptr, xh, 384, NNODE, 384, 128, 0);

    gemm(stream, ew, 480, dir0W[l], dir0b[l], nullptr, et1, 384, NEDGE, 384, 480, 1);
    gemm(stream, et1, 384, dir2W[l], dir2b[l], nullptr, et2, 384, NEDGE, 384, 384, 0);
    gemm(stream, ew + 384, 480, rbfW[l], rbfb[l], nullptr, et3, 384, NEDGE, 384, NRBF, 0);

    zero(sacc, (long)NNODE * HID);
    zero(vacc, (long)NNODE * 384);
    message_k<<<NEDGE, 128, 0, stream>>>(ei, xh, et2, et3, ediff, vecb, sacc, vacc);
    nup1_k<<<NNODE, 128, 0, stream>>>(sbuf, sacc, vecb, vacc);

    gemm(stream, vecb, 128, vpW[l], nullptr, nullptr, vpb, 256, NNODE * 3, 256, 128, 0);
    makecat_k<<<NNODE, 128, 0, stream>>>(sbuf, vpb, catb, vdotb);
    gemm(stream, catb, 256, xv0W[l], xv0b[l], nullptr, tnB, 128, NNODE, 128, 256, 1);
    gemm(stream, tnB, 128, xv2W[l], xv2b[l], nullptr, xvb, 384, NNODE, 384, 128, 0);
    nup2_k<<<NNODE, 128, 0, stream>>>(sbuf, vecb, xvb, vpb, vdotb);
  }

  // 7) readout: per-node linear + segment mean over molecules
  zero(outAcc, NMOL);
  zero(cntb, NMOL);
  out_k<<<(NNODE + 255) / 256, 256, 0, stream>>>(sbuf, lastW, lastb, batch, outAcc, cntb);
  fin_k<<<1, 64, 0, stream>>>(outAcc, cntb, (float*)d_out);
}
